// CRA_62302795596199
// MI455X (gfx1250) — compile-verified, matrix operand order checked by reading, not run on hardware
//
#include <hip/hip_runtime.h>
#include <hip/hip_bf16.h>
#include <stddef.h>

// ---------------------------------------------------------------------------
// Problem constants (match reference)
// ---------------------------------------------------------------------------
#define B_    128
#define NTOK  196          // tokens per stream
#define NT2   392          // 2*NTOK
#define CDIM  768
#define MTOT  (B_ * NTOK)  // 25088 GEMM rows per stream

// ---------------------------------------------------------------------------
// Types
// ---------------------------------------------------------------------------
typedef __bf16          bf16x16 __attribute__((ext_vector_type(16)));
typedef float           f32x8   __attribute__((ext_vector_type(8)));
typedef unsigned short  u16x8   __attribute__((ext_vector_type(8)));
typedef unsigned int    u32x4   __attribute__((ext_vector_type(4)));
typedef int             i32x8   __attribute__((ext_vector_type(8)));
typedef int             i32x4   __attribute__((ext_vector_type(4)));

union FragU {
    bf16x16 v;
    u16x8   h[2];
};

__device__ __forceinline__ unsigned short f2bf(float f) {
    unsigned u = __float_as_uint(f);
    unsigned r = u + 0x7FFFu + ((u >> 16) & 1u);   // round-to-nearest-even
    return (unsigned short)(r >> 16);
}
__device__ __forceinline__ float bf2f(unsigned short h) {
    return __uint_as_float(((unsigned)h) << 16);
}

#if __has_builtin(__builtin_amdgcn_tensor_load_to_lds)
#define USE_TDM 1
#else
#define USE_TDM 0
#endif

__device__ __forceinline__ void wait_tensorcnt0() {
#if __has_builtin(__builtin_amdgcn_s_wait_tensorcnt)
    __builtin_amdgcn_s_wait_tensorcnt(0);
#else
    asm volatile("s_wait_tensorcnt 0x0" ::: "memory");
#endif
}

#if USE_TDM
// ---------------------------------------------------------------------------
// TDM: 2-D tile load (bf16 elements) with LDS row padding.
// Tile = tile_d1 rows x tile_d0 elems, global row stride = stride_d0 elems.
// LDS padding: pad_interval=3 (16 DWORDs = 64B = one 32-elem bf16 row),
//              pad_amount=3 (4 DWORDs = 16B) -> LDS row pitch 80B (= LDA=40 u16)
// D# packing per CDNA5 ISA §8.3-8.4. 6-arg builtin (clang-23 form):
//   (u32x4 g0, i32x8 g1, i32x4 g2, i32x4 g3, i32x8 extra, i32 cpol)
// ---------------------------------------------------------------------------
__device__ __forceinline__ void tdm_load_tile_bf16(
        const unsigned short* gptr,   // tile origin in global memory
        unsigned lds_addr,            // LDS byte address of tile start
        unsigned tensor_d0, unsigned tensor_d1,
        unsigned stride_d0,           // elements
        unsigned tile_d0, unsigned tile_d1) {
    unsigned long long ga = (unsigned long long)(size_t)gptr;
    u32x4 g0;
    g0[0] = 1u;                                        // count=1, user descriptor
    g0[1] = lds_addr;                                  // lds_addr [63:32]
    g0[2] = (unsigned)(ga & 0xFFFFFFFFu);              // global_addr [95:64]
    g0[3] = (unsigned)((ga >> 32) & 0x01FFFFFFu)       // global_addr [120:96]
          | (2u << 30);                                // type=2 ("image")
    i32x8 g1;
    g1[0] = (int)((1u << 16)                           // data_size = 2 bytes
                | (1u << 20)                           // pad_enable
                | (3u << 22)                           // pad_interval: 16 DWORDs
                | (3u << 25));                         // pad_amount : 4 DWORDs
    g1[1] = (int)(tensor_d0 << 16);                    // tensor_dim0[15:0]
    g1[2] = (int)((tensor_d0 >> 16) | (tensor_d1 << 16));
    g1[3] = (int)((tensor_d1 >> 16) | (tile_d0 << 16));// tile_dim0
    g1[4] = (int)(tile_d1 & 0xFFFFu);                  // tile_dim1 (tile_dim2=0)
    g1[5] = (int)stride_d0;                            // tensor_dim0_stride[31:0]
    g1[6] = 0;
    g1[7] = 0;
    i32x4 z4 = {0, 0, 0, 0};
    i32x8 z8 = {0, 0, 0, 0, 0, 0, 0, 0};
    __builtin_amdgcn_tensor_load_to_lds(g0, g1, z4, z4, z8, 0);
}
#endif

// ---------------------------------------------------------------------------
// Kernel 0: batch-independent collapse vectors (A is symmetric => R=[A|A];
// everything downstream of the GEMMs reduces to matvecs)
//   u1[d] = sum_k W3[d,k]*W5[k]
//   q[j]  = sum_m (W4[j,m]+W4[392+j,m])*W5[768+m]
//   c     = b3.W5a + b4.W5b + b5
// ---------------------------------------------------------------------------
__global__ void cra_precompute(const float* __restrict__ W3, const float* __restrict__ b3,
                               const float* __restrict__ W4, const float* __restrict__ b4,
                               const float* __restrict__ W5, const float* __restrict__ b5,
                               float* __restrict__ u1, float* __restrict__ q,
                               float* __restrict__ cc) {
    int t = blockIdx.x * blockDim.x + threadIdx.x;
    if (t < CDIM) {
        float a = 0.f;
        for (int k = 0; k < CDIM; ++k) a += W3[(size_t)t * CDIM + k] * W5[k];
        u1[t] = a;
    } else if (t < CDIM + NT2) {
        int j = t - CDIM;
        float a = 0.f;
        for (int m = 0; m < 784; ++m)
            a += (W4[(size_t)j * 784 + m] + W4[(size_t)(NT2 + j) * 784 + m]) * W5[CDIM + m];
        q[j] = a;
    } else if (t == CDIM + NT2) {
        float a = b5[0];
        for (int k = 0; k < CDIM; ++k) a += b3[k] * W5[k];
        for (int m = 0; m < 784; ++m) a += b4[m] * W5[CDIM + m];
        *cc = a;
    }
}

// ---------------------------------------------------------------------------
// Pre-conversion kernels (one-time; remove all cvt VALU from the GEMM loop)
// ---------------------------------------------------------------------------
__global__ __launch_bounds__(256)
void cra_cvt_bf16(const float* __restrict__ src, unsigned* __restrict__ dst, int npairs) {
    int i = blockIdx.x * 256 + threadIdx.x;
    if (i < npairs) {
        float2 v = ((const float2*)src)[i];
        dst[i] = (unsigned)f2bf(v.x) | ((unsigned)f2bf(v.y) << 16);
    }
}

// Wt[c,d] = bf16(W[d,c])  (transpose so B tiles are row-major [n][k])
__global__ __launch_bounds__(256)
void cra_wt_bf16(const float* __restrict__ W, unsigned short* __restrict__ Wt) {
    int i = blockIdx.x * 256 + threadIdx.x;        // < 768*768
    int c = i / CDIM, d = i - c * CDIM;
    Wt[i] = f2bf(W[(size_t)d * CDIM + c]);
}

// ---------------------------------------------------------------------------
// Kernel 1: phi GEMM  [MTOT,768](bf16) x Wt^T -> bias -> BN affine -> ReLU
//   bf16 WMMA 16x16x32; block tile 128x128; 8 waves (4x2); wave tile 32x64.
//   Double-buffered LDS tiles filled by the Tensor Data Mover (TDM).
// ---------------------------------------------------------------------------
#define KSTEPS (CDIM / 32)   // 24
#define LDA    40            // LDS row pitch in u16 (32 data + 8 pad = 80B)

__global__ __launch_bounds__(256)
void cra_phi_gemm(const unsigned short* __restrict__ Xb,   // bf16 [MTOT,768]
                  const unsigned short* __restrict__ Wt,   // bf16 [768(c),768(d)]
                  const float* __restrict__ bias,
                  const float* __restrict__ gain,
                  const float* __restrict__ beta,
                  unsigned short* __restrict__ phi_out,    // bf16 [B,392,768]
                  int stream_off) {
    __shared__ __align__(16) unsigned short As[2][128][LDA];  // [m][k]
    __shared__ __align__(16) unsigned short Bs[2][128][LDA];  // [n][k]

    const int tid   = threadIdx.x;
    const int lane  = tid & 31;
    const int wid   = tid >> 5;
    const int waveM = wid & 3;
    const int waveN = wid >> 2;
    const int lrow  = lane & 15;
    const int khalf = lane >> 4;      // ISA 16-bit operand lane split
    const int koff  = khalf * 8;

    const int m0 = blockIdx.x * 128;
    const int n0 = blockIdx.y * 128;

    f32x8 acc[2][4];
#pragma unroll
    for (int i = 0; i < 2; ++i)
#pragma unroll
        for (int j = 0; j < 4; ++j) acc[i][j] = 0.0f;

#if USE_TDM
    auto issue = [&](int buf, int kt) {
        if (tid == 0) {   // one TDM issue per tile (EXEC ignored by TENSOR ops)
            tdm_load_tile_bf16(Xb + (size_t)m0 * CDIM + kt * 32,
                               (unsigned)(size_t)&As[buf][0][0],
                               CDIM, 128, CDIM, 32, 128);
            tdm_load_tile_bf16(Wt + (size_t)n0 * CDIM + kt * 32,
                               (unsigned)(size_t)&Bs[buf][0][0],
                               CDIM, 128, CDIM, 32, 128);
        }
    };
    auto drain = [&]() { if (tid < 32) wait_tensorcnt0(); };
#else
    auto issue = [&](int buf, int kt) {
#pragma unroll
        for (int i = 0; i < 2; ++i) {
            int id  = tid + i * 256;          // 512 x 16B segments per tile pair
            int row = id >> 2, seg = (id & 3) << 3;
            *(u16x8*)&As[buf][row][seg] =
                *(const u16x8*)(Xb + (size_t)(m0 + row) * CDIM + kt * 32 + seg);
            *(u16x8*)&Bs[buf][row][seg] =
                *(const u16x8*)(Wt + (size_t)(n0 + row) * CDIM + kt * 32 + seg);
        }
    };
    auto drain = [&]() {};
#endif

    issue(0, 0);
    drain();
    __syncthreads();

#pragma unroll 1
    for (int kt = 0; kt < KSTEPS; ++kt) {
        const int cur = kt & 1;
        if (kt + 1 < KSTEPS) issue(cur ^ 1, kt + 1);   // overlap DMA with math

        // ISA 7.12.2 16-bit fragment layout: lane<16 -> K {koff..+7, koff+16..+23}
        FragU a[2], b[4];
#pragma unroll
        for (int mf = 0; mf < 2; ++mf) {
            const int row = waveM * 32 + mf * 16 + lrow;
            a[mf].h[0] = *(const u16x8*)&As[cur][row][koff];
            a[mf].h[1] = *(const u16x8*)&As[cur][row][koff + 16];
        }
#pragma unroll
        for (int nf = 0; nf < 4; ++nf) {
            const int col = waveN * 64 + nf * 16 + lrow;
            b[nf].h[0] = *(const u16x8*)&Bs[cur][col][koff];
            b[nf].h[1] = *(const u16x8*)&Bs[cur][col][koff + 16];
        }
#pragma unroll
        for (int mf = 0; mf < 2; ++mf)
#pragma unroll
            for (int nf = 0; nf < 4; ++nf)
                acc[mf][nf] = __builtin_amdgcn_wmma_f32_16x16x32_bf16(
                    false, a[mf].v, false, b[nf].v, (short)0, acc[mf][nf], false, false);

        if (kt + 1 < KSTEPS) drain();
        __syncthreads();
    }

    // Epilogue: (acc + b)*g + be, ReLU, bf16, scatter to cat_phi[b, off+n, c]
#pragma unroll
    for (int nf = 0; nf < 4; ++nf) {
        const int c  = n0 + waveN * 64 + nf * 16 + lrow;
        const float bb = bias[c], gg = gain[c], be = beta[c];
#pragma unroll
        for (int mf = 0; mf < 2; ++mf) {
            const int mbase = m0 + waveM * 32 + mf * 16 + khalf * 8;
#pragma unroll
            for (int r = 0; r < 8; ++r) {
                const int m = mbase + r;
                const int bidx = m / NTOK;
                const int nidx = m - bidx * NTOK;
                float v = (acc[mf][nf][r] + bb) * gg + be;
                v = v > 0.f ? v : 0.f;
                phi_out[((size_t)bidx * NT2 + stream_off + nidx) * CDIM + c] = f2bf(v);
            }
        }
    }
}

// ---------------------------------------------------------------------------
// Kernel 2: wv[b,d] = u1[d] + sum_i phi[b,i,d] * q[i]   (phi^T @ q per batch)
// ---------------------------------------------------------------------------
__global__ __launch_bounds__(256)
void cra_colreduce(const unsigned short* __restrict__ phi, const float* __restrict__ q,
                   const float* __restrict__ u1, float* __restrict__ wv) {
    __shared__ float qs[NT2];
    const int b = blockIdx.x;
    const int d = blockIdx.y * 256 + threadIdx.x;
    for (int i = threadIdx.x; i < NT2; i += 256) qs[i] = q[i];
    __syncthreads();
    const unsigned short* p = phi + (size_t)b * NT2 * CDIM + d;
    float acc = 0.f;
#pragma unroll 4
    for (int i = 0; i < NT2; ++i) acc += bf2f(p[(size_t)i * CDIM]) * qs[i];
    wv[b * CDIM + d] = u1[d] + acc;
}

// ---------------------------------------------------------------------------
// Kernel 3: s[b,i] = phi[b,i,:] . wv[b,:] + c    (one wave32 per row)
// ---------------------------------------------------------------------------
__global__ __launch_bounds__(256)
void cra_rowdot(const unsigned short* __restrict__ phi, const float* __restrict__ wv,
                const float* __restrict__ cc, float* __restrict__ s) {
    const int wid  = threadIdx.x >> 5;
    const int lane = threadIdx.x & 31;
    const int row  = blockIdx.x * 8 + wid;
    const int b    = row / NT2;
    const unsigned short* p = phi + (size_t)row * CDIM;
    const float* w = wv + (size_t)b * CDIM;
    float acc = 0.f;
#pragma unroll
    for (int j = 0; j < CDIM / 64; ++j) {
        const int d = j * 64 + lane * 2;
        const unsigned u = *(const unsigned*)(p + d);
        acc += bf2f((unsigned short)u) * w[d] + bf2f((unsigned short)(u >> 16)) * w[d + 1];
    }
#pragma unroll
    for (int off = 16; off; off >>= 1) acc += __shfl_xor(acc, off, 32);
    if (lane == 0) s[row] = acc + *cc;
}

// ---------------------------------------------------------------------------
// Kernel 4: out = x * s_x + y * s_y
// ---------------------------------------------------------------------------
__global__ __launch_bounds__(256)
void cra_combine(const float* __restrict__ x, const float* __restrict__ y,
                 const float* __restrict__ s, float* __restrict__ out) {
    const int idx = blockIdx.x * 256 + threadIdx.x;
    const int b = idx / (NTOK * CDIM);
    const int r = idx - b * (NTOK * CDIM);
    const int n = r / CDIM;
    out[idx] = x[idx] * s[b * NT2 + n] + y[idx] * s[b * NT2 + NTOK + n];
}

// ---------------------------------------------------------------------------
// Launch
// ---------------------------------------------------------------------------
extern "C" void kernel_launch(void* const* d_in, const int* in_sizes, int n_in,
                              void* d_out, int out_size, void* d_ws, size_t ws_size,
                              hipStream_t stream) {
    const float* x  = (const float*)d_in[0];
    const float* y  = (const float*)d_in[1];
    const float* W1 = (const float*)d_in[2];
    const float* b1 = (const float*)d_in[3];
    const float* g1 = (const float*)d_in[4];
    const float* be1= (const float*)d_in[5];
    const float* W2 = (const float*)d_in[6];
    const float* b2 = (const float*)d_in[7];
    const float* g2 = (const float*)d_in[8];
    const float* be2= (const float*)d_in[9];
    const float* W3 = (const float*)d_in[10];
    const float* b3 = (const float*)d_in[11];
    const float* W4 = (const float*)d_in[12];
    const float* b4 = (const float*)d_in[13];
    const float* W5 = (const float*)d_in[14];
    const float* b5 = (const float*)d_in[15];
    float* out = (float*)d_out;

    // workspace carve-up
    char* ws = (char*)d_ws;
    size_t off = 0;
    unsigned short* phi = (unsigned short*)(ws + off); off += (size_t)B_ * NT2 * CDIM * 2;
    unsigned short* Xb  = (unsigned short*)(ws + off); off += (size_t)MTOT * CDIM * 2;   // reused x then y
    unsigned short* Wt  = (unsigned short*)(ws + off); off += (size_t)CDIM * CDIM * 2;   // reused W1 then W2
    float* u1 = (float*)(ws + off); off += CDIM * sizeof(float);
    float* q  = (float*)(ws + off); off += NT2 * sizeof(float);
    float* cc = (float*)(ws + off); off += 256;
    float* wv = (float*)(ws + off); off += (size_t)B_ * CDIM * sizeof(float);
    float* s  = (float*)(ws + off); off += (size_t)B_ * NT2 * sizeof(float);

    const int npairs = MTOT * CDIM / 2;
    const dim3 gg(MTOT / 128, CDIM / 128);   // (196, 6)

    // 0) collapse vectors
    cra_precompute<<<5, 256, 0, stream>>>(W3, b3, W4, b4, W5, b5, u1, q, cc);

    // 1) stream x:  f32->bf16, W1^T->bf16, GEMM -> cat_phi[:, :196, :]
    cra_cvt_bf16<<<(npairs + 255) / 256, 256, 0, stream>>>(x, (unsigned*)Xb, npairs);
    cra_wt_bf16 <<<(CDIM * CDIM) / 256, 256, 0, stream>>>(W1, Wt);
    cra_phi_gemm<<<gg, 256, 0, stream>>>(Xb, Wt, b1, g1, be1, phi, 0);

    // 2) stream y (reuse Xb/Wt buffers)
    cra_cvt_bf16<<<(npairs + 255) / 256, 256, 0, stream>>>(y, (unsigned*)Xb, npairs);
    cra_wt_bf16 <<<(CDIM * CDIM) / 256, 256, 0, stream>>>(W2, Wt);
    cra_phi_gemm<<<gg, 256, 0, stream>>>(Xb, Wt, b2, g2, be2, phi, NTOK);

    // 3) wv = u1 + phi^T q ; s = phi.wv + c ; out = x*s_x + y*s_y
    cra_colreduce<<<dim3(B_, CDIM / 256), 256, 0, stream>>>(phi, q, u1, wv);
    cra_rowdot<<<(B_ * NT2) / 8, 256, 0, stream>>>(phi, wv, cc, s);
    cra_combine<<<(B_ * NTOK * CDIM) / 256, 256, 0, stream>>>(x, y, s, out);
}
